// SimpleRNNModel_1760936591467
// MI455X (gfx1250) — matrix operand order repaired and checked by reading, not compiled, not run on hardware
//
#include <hip/hip_runtime.h>
#include <hip/hip_bf16.h>
#include <math.h>

#define SEQ   1024
#define BATCH 64
#define ISZ   512
#define HSZ   512

typedef __attribute__((ext_vector_type(16))) __bf16 v16bf;
typedef __attribute__((ext_vector_type(8)))  float  v8f;

union Frag {
    v16bf v;
    uint4 q[2];
    unsigned short s[16];
};

// f32 -> bf16, round to nearest even
__device__ __forceinline__ unsigned short f2bf(float f) {
    unsigned u = __float_as_uint(f);
    unsigned r = u + 0x7FFFu + ((u >> 16) & 1u);
    return (unsigned short)(r >> 16);
}

// ---------------------------------------------------------------------------
// Kernel 1: convert W (512 x 1024, [Wh | Wx]) to bf16 halves, zero h ping-pong
// buffers, reset the grid-barrier counter (deterministic per launch).
// ---------------------------------------------------------------------------
__global__ void rnn_prep(const float* __restrict__ W,
                         unsigned short* __restrict__ Whbf,
                         unsigned short* __restrict__ Wxbf,
                         unsigned short* __restrict__ hA,
                         unsigned short* __restrict__ hB,
                         unsigned* __restrict__ cnt) {
    int tid = blockIdx.x * blockDim.x + threadIdx.x;
    const int total = HSZ * (ISZ + HSZ);
    for (int e = tid; e < total; e += gridDim.x * blockDim.x) {
        int j = e >> 10;          // row of W
        int c = e & 1023;         // col of W
        unsigned short bv = f2bf(__builtin_nontemporal_load(W + e));
        if (c < HSZ) Whbf[j * HSZ + c]         = bv;
        else         Wxbf[j * ISZ + (c - HSZ)] = bv;
    }
    if (tid < BATCH * HSZ) { hA[tid] = 0; hB[tid] = 0; }
    if (tid == 0) *cnt = 0u;
}

// ---------------------------------------------------------------------------
// Kernel 2: preact[t,b,:] = x[t,b,:] @ Wx^T + bias   (parallel over all rows)
// Block = 1024 threads (32 waves). Block owns 16 rows of X (staged once into
// LDS as bf16, non-temporal read: X has zero reuse); wave w owns hidden column
// tile w. 16 x WMMA bf16 K-slices. preact stored non-temporal (written once,
// read once much later) to keep L2 for Wx / h.
// ---------------------------------------------------------------------------
__global__ __launch_bounds__(1024) void rnn_preact(
    const float* __restrict__ X,               // [SEQ*BATCH, ISZ] f32
    const unsigned short* __restrict__ Wxbf,   // [HSZ, ISZ] bf16
    const float* __restrict__ bias,            // [HSZ]
    float* __restrict__ preact)                // [SEQ*BATCH, HSZ] f32
{
    __shared__ __align__(16) unsigned short sh[16 * ISZ];   // 16 KB
    const int rt   = blockIdx.x;               // row tile (16 rows)
    const int lane = threadIdx.x & 31;
    const int w    = threadIdx.x >> 5;         // column tile 0..31

    // Stage 16 rows of X into LDS as bf16 (each element of X read exactly once).
    const float* xrow = X + (size_t)rt * 16 * ISZ;
#pragma unroll
    for (int i = 0; i < 8; ++i) {
        int e = threadIdx.x + 1024 * i;        // 0..8191
        sh[e] = f2bf(__builtin_nontemporal_load(xrow + e));
    }
    __syncthreads();

    const int m   = lane & 15;                 // A row / B col within tile
    const int kbA = (lane < 16) ? 0 : 8;       // A-frag K interleave
    const int koB = (lane < 16) ? 0 : 8;       // B-frag K offset
    const unsigned short* brow = Wxbf + (size_t)(16 * w + m) * ISZ;

    v8f acc = {};
#pragma unroll
    for (int kk = 0; kk < ISZ / 32; ++kk) {
        const int k0 = kk * 32;
        Frag a, b;
        a.q[0] = *(const uint4*)(sh + m * ISZ + k0 + kbA);
        a.q[1] = *(const uint4*)(sh + m * ISZ + k0 + 16 + kbA);
        b.q[0] = *(const uint4*)(brow + k0 + koB);
        b.q[1] = *(const uint4*)(brow + k0 + koB + 16);
        acc = __builtin_amdgcn_wmma_f32_16x16x32_bf16(
                  false, a.v, false, b.v, (short)0, acc, false, false);
    }

    const int mo = (lane < 16) ? 0 : 8;
    float* crow = preact + (size_t)rt * 16 * HSZ;
    const float bb = bias[16 * w + m];
#pragma unroll
    for (int r = 0; r < 8; ++r) {
        int mm  = r + mo;
        int col = 16 * w + m;
        __builtin_nontemporal_store(acc[r] + bb, crow + (size_t)mm * HSZ + col);
    }
}

// ---------------------------------------------------------------------------
// Kernel 3: persistent recurrence. 16 blocks x 256 threads = 128 waves = one
// 16x16 output tile per wave. Wh lives in registers as 16 prebuilt WMMA
// B-fragments (loaded once, zero weight traffic in the loop). h ping-pongs
// through L2 in bf16. preact tile for step t is loaded before the WMMA chain
// (overlaps matrix ops); step t+1's tile is prefetched (global_prefetch) just
// before the barrier spin, so its HBM latency hides behind the device-wide
// sync instead of sitting on the serial critical path.
// ---------------------------------------------------------------------------
__global__ __launch_bounds__(256, 1) void rnn_recur(
    const unsigned short* __restrict__ Whbf,   // [HSZ, HSZ] bf16
    const float* __restrict__ preact,          // [SEQ*BATCH, HSZ] f32
    unsigned short* __restrict__ hA,           // [BATCH, HSZ] bf16
    unsigned short* __restrict__ hB,           // [BATCH, HSZ] bf16
    float* __restrict__ out,                   // [BATCH, HSZ] f32
    unsigned* __restrict__ cnt)
{
    const int lane = threadIdx.x & 31;
    const int wave = threadIdx.x >> 5;
    const int gid  = blockIdx.x * 8 + wave;    // 0..127
    const int mr   = gid & 3;                  // batch row-tile (4)
    const int nc   = gid >> 2;                 // hidden col-tile (32)

    const int n    = lane & 15;
    const int kbA  = (lane < 16) ? 0 : 8;
    const int koB  = (lane < 16) ? 0 : 8;
    const int mo   = (lane < 16) ? 0 : 8;

    // Preload this column tile's Wh B-fragments: 16 K-slices, 128 VGPRs.
    Frag wh[16];
    const unsigned short* bhrow = Whbf + (size_t)(16 * nc + n) * HSZ;
#pragma unroll
    for (int kk = 0; kk < 16; ++kk) {
        wh[kk].q[0] = *(const uint4*)(bhrow + 32 * kk + koB);
        wh[kk].q[1] = *(const uint4*)(bhrow + 32 * kk + koB + 16);
    }

    // Warm L2 with step 0's preact tile.
    {
        const float* p0 = preact;
#pragma unroll
        for (int r = 0; r < 8; ++r)
            __builtin_prefetch(p0 + (size_t)(16 * mr + r + mo) * HSZ + 16 * nc + n, 0, 3);
    }

    unsigned short* bufs[2] = { hA, hB };
    const unsigned nwg = gridDim.x;

#pragma unroll 1
    for (int t = 0; t < SEQ; ++t) {
        const unsigned short* cur = bufs[t & 1];
        unsigned short*       nxt = bufs[(t + 1) & 1];
        const float* prow = preact + (size_t)t * BATCH * HSZ;

        // Issue the preact tile loads first so they overlap the WMMA chain.
        float pv[8];
#pragma unroll
        for (int r = 0; r < 8; ++r)
            pv[r] = prow[(size_t)(16 * mr + r + mo) * HSZ + 16 * nc + n];

        v8f acc = {};
        const unsigned short* arow = cur + (size_t)(16 * mr + n) * HSZ;
#pragma unroll
        for (int kk = 0; kk < 16; ++kk) {
            Frag a;
            a.q[0] = *(const uint4*)(arow + 32 * kk + kbA);
            a.q[1] = *(const uint4*)(arow + 32 * kk + 16 + kbA);
            acc = __builtin_amdgcn_wmma_f32_16x16x32_bf16(
                      false, a.v, false, wh[kk].v, (short)0, acc, false, false);
        }

        const bool last = (t == SEQ - 1);
#pragma unroll
        for (int r = 0; r < 8; ++r) {
            int b_row = 16 * mr + r + mo;
            int col   = 16 * nc + n;
            float v = tanhf(acc[r] + pv[r]);
            nxt[(size_t)b_row * HSZ + col] = f2bf(v);
            if (last) out[(size_t)b_row * HSZ + col] = v;
        }

        // Prefetch next step's preact tile; its HBM latency hides behind the
        // barrier spin instead of serializing into step t+1.
        if (t + 1 < SEQ) {
            const float* pnext = prow + BATCH * HSZ;
#pragma unroll
            for (int r = 0; r < 8; ++r)
                __builtin_prefetch(pnext + (size_t)(16 * mr + r + mo) * HSZ + 16 * nc + n, 0, 3);
        }

        // -------- device-wide step barrier --------
        __threadfence();
        __syncthreads();
        if (threadIdx.x == 0) {
            atomicAdd(cnt, 1u);
            const unsigned tgt = (unsigned)(t + 1) * nwg;
            while (atomicAdd(cnt, 0u) < tgt) {
                __builtin_amdgcn_s_sleep(1);
            }
        }
        __syncthreads();
        __threadfence();
    }
}

// ---------------------------------------------------------------------------
extern "C" void kernel_launch(void* const* d_in, const int* in_sizes, int n_in,
                              void* d_out, int out_size, void* d_ws, size_t ws_size,
                              hipStream_t stream) {
    (void)in_sizes; (void)n_in; (void)out_size; (void)ws_size;
    const float* X = (const float*)d_in[0];   // [1024, 64, 512]
    const float* W = (const float*)d_in[1];   // [512, 1024]
    const float* b = (const float*)d_in[2];   // [512]
    float* out = (float*)d_out;               // [64, 512]

    char* ws = (char*)d_ws;
    size_t off = 0;
    unsigned short* Whbf = (unsigned short*)(ws + off); off += (size_t)HSZ * HSZ * 2;
    unsigned short* Wxbf = (unsigned short*)(ws + off); off += (size_t)HSZ * ISZ * 2;
    unsigned short* hA   = (unsigned short*)(ws + off); off += (size_t)BATCH * HSZ * 2;
    unsigned short* hB   = (unsigned short*)(ws + off); off += (size_t)BATCH * HSZ * 2;
    unsigned*       cnt  = (unsigned*)(ws + off);       off += 256;
    float*          pre  = (float*)(ws + off);          off += (size_t)SEQ * BATCH * HSZ * 4;

    rnn_prep  <<<2048, 256, 0, stream>>>(W, Whbf, Wxbf, hA, hB, cnt);
    rnn_preact<<<(SEQ * BATCH) / 16, 1024, 0, stream>>>(X, Wxbf, b, pre);
    rnn_recur <<<16, 256, 0, stream>>>(Whbf, pre, hA, hB, out, cnt);
}
